// AdvancedStockRNN_88038239634097
// MI455X (gfx1250) — compile-verified, matrix operand order checked by reading, not run on hardware
//
#include <hip/hip_runtime.h>
#include <hip/hip_bf16.h>

#define BB  256
#define TT  512
#define DINW 64
#define HHW 256
#define HBW 128
#define BTN (BB * TT)

typedef unsigned short u16;
typedef __attribute__((ext_vector_type(16))) __bf16 v16bf;
typedef __attribute__((ext_vector_type(8)))  float  v8f;

union BFrag { v16bf v; u16 u[16]; };

__device__ __forceinline__ u16 f2bf(float f) {
  unsigned int u = __float_as_uint(f);
  u += 0x7FFFu + ((u >> 16) & 1u);          // round-to-nearest-even
  return (u16)(u >> 16);
}
__device__ __forceinline__ float bf2f(u16 u) {
  return __uint_as_float(((unsigned int)u) << 16);
}

__device__ __forceinline__ v8f wmma_bf16(v16bf a, v16bf b, v8f c) {
  // D = A(16x32 bf16) * B(32x16 bf16) + C(16x16 f32)
  return __builtin_amdgcn_wmma_f32_16x16x32_bf16(false, a, false, b, (short)0, c,
                                                 false, false);
}

// 16x32 bf16 fragment from a row-major [rows, K] bf16 source. Two contiguous
// 16B chunks per lane -> two global_load_b128, zero conversion VALU.
// Lane l<16: row=l,   elems 0..7 -> K=k0..k0+7,   elems 8..15 -> K=k0+16..k0+23
// Lane l>=16:row=l-16,elems 0..7 -> K=k0+8..k0+15,elems 8..15 -> K=k0+24..k0+31
__device__ __forceinline__ v16bf ldtile_bf(const u16* base, long ld, int k0) {
  int lane = threadIdx.x & 31;
  const u16* p = base + (long)(lane & 15) * ld + k0 + ((lane >> 4) << 3);
  BFrag f;
#pragma unroll
  for (int i = 0; i < 8; ++i) { f.u[i] = p[i]; f.u[8 + i] = p[16 + i]; }
  return f.v;
}

__device__ __forceinline__ float sigmoidf_(float x) { return 1.f / (1.f + expf(-x)); }

// ---------------------------------------------------------------------------
// Generic GEMM:  C[M,N] = act( A[M,K] * W[N,K]^T + bias[N] )
// A and W are bf16; C is f32 (storebf=0) or bf16 (storebf=1).
// grid = (M/64, N/64), block = 128 (4 waves, each 32x32 tile => 2x2 WMMA)
// act: 0 = none, 1 = tanh, 2 = relu
// ---------------------------------------------------------------------------
__global__ void gemm_wmma(const u16* __restrict__ A, long lda,
                          const u16* __restrict__ W, long ldw,
                          const float* __restrict__ bias,
                          void* __restrict__ C, long ldc, int K, int act,
                          int storebf) {
  int wave = threadIdx.x >> 5;
  int m0 = blockIdx.x * 64 + (wave >> 1) * 32;
  int n0 = blockIdx.y * 64 + (wave & 1) * 32;
  const u16* A0 = A + (long)m0 * lda;
  const u16* A1 = A + (long)(m0 + 16) * lda;
  const u16* W0 = W + (long)n0 * ldw;
  const u16* W1 = W + (long)(n0 + 16) * ldw;
  v8f a00 = {}, a01 = {}, a10 = {}, a11 = {};
  for (int k = 0; k < K; k += 32) {
    if (k + 32 < K) {  // global_prefetch_b8 next K panel
      __builtin_prefetch(A0 + (long)(threadIdx.x & 15) * lda + k + 32, 0, 0);
      __builtin_prefetch(A1 + (long)(threadIdx.x & 15) * lda + k + 32, 0, 0);
    }
    v16bf fa0 = ldtile_bf(A0, lda, k);
    v16bf fa1 = ldtile_bf(A1, lda, k);
    v16bf fb0 = ldtile_bf(W0, ldw, k);
    v16bf fb1 = ldtile_bf(W1, ldw, k);
    a00 = wmma_bf16(fa0, fb0, a00);
    a01 = wmma_bf16(fa0, fb1, a01);
    a10 = wmma_bf16(fa1, fb0, a10);
    a11 = wmma_bf16(fa1, fb1, a11);
  }
  int lane = threadIdx.x & 31;
  int col = lane & 15;
  int rof = (lane >> 4) << 3;
  const v8f* accs[4] = {&a00, &a01, &a10, &a11};
#pragma unroll
  for (int mi = 0; mi < 2; ++mi)
#pragma unroll
    for (int ni = 0; ni < 2; ++ni) {
      const v8f& a = *accs[mi * 2 + ni];
#pragma unroll
      for (int j = 0; j < 8; ++j) {
        int m = m0 + mi * 16 + rof + j;
        int n = n0 + ni * 16 + col;
        float v = a[j] + (bias ? bias[n] : 0.f);
        if (act == 1) v = tanhf(v);
        else if (act == 2) v = fmaxf(v, 0.f);
        if (storebf) ((u16*)C)[(long)m * ldc + n] = f2bf(v);
        else ((float*)C)[(long)m * ldc + n] = v;
      }
    }
}

// ---------------------------------------------------------------------------
// One tanh-RNN timestep: out = tanh(xw + h_prev * Whh^T + bhh)
// h buffers are bf16 [B,T,H]; A rows indexed by batch b with stride lda = T*H.
// ---------------------------------------------------------------------------
__global__ void rnn_step(const u16* __restrict__ hprev, long lda, int has_prev,
                         const u16* __restrict__ Whh, long ldw,
                         const float* __restrict__ bhh,
                         const float* __restrict__ xw, long ldx,
                         u16* __restrict__ out, long ldo, int K) {
  int wave = threadIdx.x >> 5;
  int m0 = blockIdx.x * 64 + (wave >> 1) * 32;
  int n0 = blockIdx.y * 64 + (wave & 1) * 32;
  v8f a00 = {}, a01 = {}, a10 = {}, a11 = {};
  if (has_prev) {
    const u16* A0 = hprev + (long)m0 * lda;
    const u16* A1 = hprev + (long)(m0 + 16) * lda;
    const u16* W0 = Whh + (long)n0 * ldw;
    const u16* W1 = Whh + (long)(n0 + 16) * ldw;
    for (int k = 0; k < K; k += 32) {
      v16bf fa0 = ldtile_bf(A0, lda, k);
      v16bf fa1 = ldtile_bf(A1, lda, k);
      v16bf fb0 = ldtile_bf(W0, ldw, k);
      v16bf fb1 = ldtile_bf(W1, ldw, k);
      a00 = wmma_bf16(fa0, fb0, a00);
      a01 = wmma_bf16(fa0, fb1, a01);
      a10 = wmma_bf16(fa1, fb0, a10);
      a11 = wmma_bf16(fa1, fb1, a11);
    }
  }
  int lane = threadIdx.x & 31;
  int col = lane & 15;
  int rof = (lane >> 4) << 3;
  const v8f* accs[4] = {&a00, &a01, &a10, &a11};
#pragma unroll
  for (int mi = 0; mi < 2; ++mi)
#pragma unroll
    for (int ni = 0; ni < 2; ++ni) {
      const v8f& a = *accs[mi * 2 + ni];
#pragma unroll
      for (int j = 0; j < 8; ++j) {
        int m = m0 + mi * 16 + rof + j;
        int n = n0 + ni * 16 + col;
        float v = a[j] + bhh[n] + xw[(long)m * ldx + n];
        out[(long)m * ldo + n] = f2bf(tanhf(v));
      }
    }
}

// ---------------------------------------------------------------------------
// One GRU timestep (gates r,z,n). All 3 gate GEMM slabs in registers (12 WMMA
// accumulators per wave), gate math done directly on the WMMA D layout.
// grid = (B/64, Hh/64).
// ---------------------------------------------------------------------------
__global__ void gru_step(const u16* __restrict__ hprev, long lda, int has_prev,
                         const u16* __restrict__ Whh, long ldw, int Hh,
                         const float* __restrict__ bhh,
                         const float* __restrict__ xg, long ldx,
                         u16* __restrict__ out, long ldo) {
  int wave = threadIdx.x >> 5;
  int m0 = blockIdx.x * 64 + (wave >> 1) * 32;
  int j0 = blockIdx.y * 64 + (wave & 1) * 32;
  v8f acc[3][4] = {};
  if (has_prev) {
    const u16* A0 = hprev + (long)m0 * lda;
    const u16* A1 = hprev + (long)(m0 + 16) * lda;
    for (int k = 0; k < Hh; k += 32) {
      v16bf fa0 = ldtile_bf(A0, lda, k);
      v16bf fa1 = ldtile_bf(A1, lda, k);
#pragma unroll
      for (int g = 0; g < 3; ++g) {
        const u16* Wg = Whh + (long)(g * Hh + j0) * ldw;
        v16bf fb0 = ldtile_bf(Wg, ldw, k);
        v16bf fb1 = ldtile_bf(Wg + (long)16 * ldw, ldw, k);
        acc[g][0] = wmma_bf16(fa0, fb0, acc[g][0]);
        acc[g][1] = wmma_bf16(fa0, fb1, acc[g][1]);
        acc[g][2] = wmma_bf16(fa1, fb0, acc[g][2]);
        acc[g][3] = wmma_bf16(fa1, fb1, acc[g][3]);
      }
    }
  }
  int lane = threadIdx.x & 31;
  int col = lane & 15;
  int rof = (lane >> 4) << 3;
#pragma unroll
  for (int mi = 0; mi < 2; ++mi)
#pragma unroll
    for (int ni = 0; ni < 2; ++ni) {
      int idx = mi * 2 + ni;
#pragma unroll
      for (int j = 0; j < 8; ++j) {
        int m = m0 + mi * 16 + rof + j;
        int jj = j0 + ni * 16 + col;
        float hr = acc[0][idx][j] + bhh[jj];
        float hz = acc[1][idx][j] + bhh[Hh + jj];
        float hn = acc[2][idx][j] + bhh[2 * Hh + jj];
        const float* xr = xg + (long)m * ldx;
        float r = sigmoidf_(xr[jj] + hr);
        float z = sigmoidf_(xr[Hh + jj] + hz);
        float nn = tanhf(xr[2 * Hh + jj] + r * hn);
        float hp = has_prev ? bf2f(hprev[(long)m * lda + jj]) : 0.f;
        out[(long)m * ldo + jj] = f2bf((1.f - z) * nn + z * hp);
      }
    }
}

// ---------------------------------------------------------------------------
// Small helper kernels
// ---------------------------------------------------------------------------
__global__ void f2bf_kernel(const float* __restrict__ src,
                            u16* __restrict__ dst, long n) {
  long i = (long)blockIdx.x * blockDim.x + threadIdx.x;
  if (i < n) dst[i] = f2bf(src[i]);
}

__global__ void score_kernel(const float* __restrict__ tmp,
                             const float* __restrict__ w2,
                             const float* __restrict__ b2,
                             float* __restrict__ sc, int n) {
  int i = blockIdx.x * 256 + threadIdx.x;
  if (i >= n) return;
  const float* p = tmp + (long)i * 128;
  float s = b2[0];
#pragma unroll 4
  for (int j = 0; j < 128; ++j) s += p[j] * w2[j];
  sc[i] = s;
}

// Softmax over T + weighted pool over T + BatchNorm(eval). One block per b.
// h is bf16; ctx written as bf16 (feeds the fc1 WMMA GEMM).
__global__ void attn_pool(const u16* __restrict__ h, const float* __restrict__ sc,
                          const float* __restrict__ gamma, const float* __restrict__ beta,
                          const float* __restrict__ mean, const float* __restrict__ var,
                          u16* __restrict__ ctx) {
  __shared__ float w[TT];
  __shared__ float red[256];
  int b = blockIdx.x, tid = threadIdx.x;
  float s0 = sc[b * TT + tid], s1 = sc[b * TT + 256 + tid];
  red[tid] = fmaxf(s0, s1);
  __syncthreads();
  for (int s = 128; s > 0; s >>= 1) {
    if (tid < s) red[tid] = fmaxf(red[tid], red[tid + s]);
    __syncthreads();
  }
  float mx = red[0];
  __syncthreads();
  float e0 = expf(s0 - mx), e1 = expf(s1 - mx);
  w[tid] = e0; w[256 + tid] = e1;
  red[tid] = e0 + e1;
  __syncthreads();
  for (int s = 128; s > 0; s >>= 1) {
    if (tid < s) red[tid] += red[tid + s];
    __syncthreads();
  }
  float inv = 1.f / red[0];
  float a = 0.f;
  const u16* hb = h + (long)b * TT * HHW + tid;
  for (int t = 0; t < TT; ++t) a += w[t] * bf2f(hb[(long)t * HHW]);
  a *= inv;
  a = (a - mean[tid]) * rsqrtf(var[tid] + 1e-5f) * gamma[tid] + beta[tid];
  ctx[b * HHW + tid] = f2bf(a);
}

__global__ void fc3_kernel(const float* __restrict__ o2, const float* __restrict__ w3,
                           const float* __restrict__ b3, float* __restrict__ out) {
  int b = blockIdx.x * blockDim.x + threadIdx.x;
  if (b >= BB) return;
  const float* p = o2 + b * 64;
  float s = b3[0];
#pragma unroll 4
  for (int j = 0; j < 64; ++j) s += p[j] * w3[j];
  out[b] = s;
}

// ---------------------------------------------------------------------------
extern "C" void kernel_launch(void* const* d_in, const int* in_sizes, int n_in,
                              void* d_out, int out_size, void* d_ws, size_t ws_size,
                              hipStream_t stream) {
  (void)in_sizes; (void)n_in; (void)out_size; (void)ws_size;
  const float* x        = (const float*)d_in[0];
  const float* rnn_wih0 = (const float*)d_in[1];
  const float* rnn_whh0 = (const float*)d_in[2];
  const float* rnn_bih0 = (const float*)d_in[3];
  const float* rnn_bhh0 = (const float*)d_in[4];
  const float* rnn_wih1 = (const float*)d_in[5];
  const float* rnn_whh1 = (const float*)d_in[6];
  const float* rnn_bih1 = (const float*)d_in[7];
  const float* rnn_bhh1 = (const float*)d_in[8];
  const float* gru_wih  = (const float*)d_in[9];
  const float* gru_whh  = (const float*)d_in[10];
  const float* gru_bih  = (const float*)d_in[11];
  const float* gru_bhh  = (const float*)d_in[12];
  const float* bi_wih   = (const float*)d_in[13];
  const float* bi_whh   = (const float*)d_in[14];
  const float* bi_bih   = (const float*)d_in[15];
  const float* bi_bhh   = (const float*)d_in[16];
  const float* att_w1   = (const float*)d_in[17];
  const float* att_b1   = (const float*)d_in[18];
  const float* att_w2   = (const float*)d_in[19];
  const float* att_b2   = (const float*)d_in[20];
  const float* bn_gamma = (const float*)d_in[21];
  const float* bn_beta  = (const float*)d_in[22];
  const float* bn_mean  = (const float*)d_in[23];
  const float* bn_var   = (const float*)d_in[24];
  const float* fc_w1    = (const float*)d_in[25];
  const float* fc_b1    = (const float*)d_in[26];
  const float* fc_w2    = (const float*)d_in[27];
  const float* fc_b2    = (const float*)d_in[28];
  const float* fc_w3    = (const float*)d_in[29];
  const float* fc_b3    = (const float*)d_in[30];

  char* ws = (char*)d_ws;
  size_t cur = 0;
  auto alloc = [&](size_t bytes) {
    size_t o = cur; cur = (cur + bytes + 255) & ~(size_t)255; return o;
  };
  // bf16 activation buffers (WMMA inputs)
  u16*   hA     = (u16*)(ws + alloc((size_t)BTN * HHW * 2));
  u16*   hB     = (u16*)(ws + alloc((size_t)BTN * HHW * 2));
  u16*   xbf    = (u16*)(ws + alloc((size_t)BTN * DINW * 2));
  u16*   ctx    = (u16*)(ws + alloc((size_t)BB * HHW * 2));
  u16*   o1     = (u16*)(ws + alloc((size_t)BB * 128 * 2));
  // f32 buffers (element-wise consumers only)
  float* XG     = (float*)(ws + alloc((size_t)BTN * 768 * 4)); // gate preacts; reused for attn tmp
  float* scores = (float*)(ws + alloc((size_t)BTN * 4));
  float* o2     = (float*)(ws + alloc((size_t)BB * 64 * 4));
  // bf16 weights
  u16* c_wih0 = (u16*)(ws + alloc(2ull * 256 * 64));
  u16* c_whh0 = (u16*)(ws + alloc(2ull * 256 * 256));
  u16* c_wih1 = (u16*)(ws + alloc(2ull * 256 * 256));
  u16* c_whh1 = (u16*)(ws + alloc(2ull * 256 * 256));
  u16* c_gwih = (u16*)(ws + alloc(2ull * 2 * 768 * 256));
  u16* c_gwhh = (u16*)(ws + alloc(2ull * 2 * 768 * 256));
  u16* c_bwih = (u16*)(ws + alloc(2ull * 4 * 384 * 256));
  u16* c_bwhh = (u16*)(ws + alloc(2ull * 4 * 384 * 128));
  u16* c_aw1  = (u16*)(ws + alloc(2ull * 128 * 256));
  u16* c_f1   = (u16*)(ws + alloc(2ull * 128 * 256));
  u16* c_f2   = (u16*)(ws + alloc(2ull * 64 * 128));

  auto conv = [&](const float* s, u16* dpt, long n) {
    f2bf_kernel<<<dim3((unsigned)((n + 255) / 256)), dim3(256), 0, stream>>>(s, dpt, n);
  };
  auto gemm = [&](const u16* A, long lda, const u16* W, long ldw, const float* bias,
                  void* C, long ldc, int M, int N, int K, int act, int storebf) {
    gemm_wmma<<<dim3(M / 64, N / 64), dim3(128), 0, stream>>>(A, lda, W, ldw, bias, C,
                                                              ldc, K, act, storebf);
  };

  // --- convert x + weights to bf16 once ---
  conv(x, xbf, (long)BTN * DINW);
  conv(rnn_wih0, c_wih0, 256 * 64);
  conv(rnn_whh0, c_whh0, 256 * 256);
  conv(rnn_wih1, c_wih1, 256 * 256);
  conv(rnn_whh1, c_whh1, 256 * 256);
  conv(gru_wih, c_gwih, 2L * 768 * 256);
  conv(gru_whh, c_gwhh, 2L * 768 * 256);
  conv(bi_wih, c_bwih, 4L * 384 * 256);
  conv(bi_whh, c_bwhh, 4L * 384 * 128);
  conv(att_w1, c_aw1, 128 * 256);
  conv(fc_w1, c_f1, 128 * 256);
  conv(fc_w2, c_f2, 64 * 128);

  const long ldh = (long)TT * HHW;

  // --- tanh-RNN layer: big input GEMM, then 512 fused WMMA step kernels ---
  auto rnn_layer_run = [&](const u16* in, int inW, const u16* wih_bf, const float* bih,
                           const u16* whh_bf, const float* bhh, u16* outbuf) {
    gemm(in, inW, wih_bf, inW, bih, XG, HHW, BTN, HHW, inW, 0, 0);
    for (int t = 0; t < TT; ++t) {
      rnn_step<<<dim3(BB / 64, HHW / 64), dim3(128), 0, stream>>>(
          outbuf + (long)(t - 1) * HHW, ldh, (t > 0) ? 1 : 0, whh_bf, HHW, bhh,
          XG + (long)t * HHW, ldh, outbuf + (long)t * HHW, ldh, HHW);
    }
  };

  // --- GRU direction: input GEMM for all gates, then fused step kernels ---
  auto gru_dir = [&](const u16* in, const u16* wih_bf, const float* bih,
                     const u16* whh_bf, const float* bhh, int Hh, u16* outbuf,
                     int colOff, int reverse) {
    gemm(in, HHW, wih_bf, HHW, bih, XG, 3 * Hh, BTN, 3 * Hh, HHW, 0, 0);
    for (int i = 0; i < TT; ++i) {
      int t = reverse ? (TT - 1 - i) : i;
      int tp = reverse ? (t + 1) : (t - 1);
      gru_step<<<dim3(BB / 64, Hh / 64), dim3(128), 0, stream>>>(
          outbuf + (long)tp * HHW + colOff, ldh, (i > 0) ? 1 : 0, whh_bf, Hh, Hh, bhh,
          XG + (long)t * 3 * Hh, (long)TT * 3 * Hh, outbuf + (long)t * HHW + colOff, ldh);
    }
  };

  // stacked tanh-RNN (2 layers)
  rnn_layer_run(xbf, DINW, c_wih0, rnn_bih0, c_whh0, rnn_bhh0, hA);
  rnn_layer_run(hA, HHW, c_wih1, rnn_bih1, c_whh1, rnn_bhh1, hB);
  // stacked GRU (2 layers)
  gru_dir(hB, c_gwih, gru_bih, c_gwhh, gru_bhh, HHW, hA, 0, 0);
  gru_dir(hA, c_gwih + 768L * 256, gru_bih + 768, c_gwhh + 768L * 256, gru_bhh + 768,
          HHW, hB, 0, 0);
  // bi-GRU layer 0: fwd -> hA[:, :128], bwd -> hA[:, 128:]
  gru_dir(hB, c_bwih, bi_bih, c_bwhh, bi_bhh, HBW, hA, 0, 0);
  gru_dir(hB, c_bwih + 384L * 256, bi_bih + 384, c_bwhh + 384L * 128, bi_bhh + 384,
          HBW, hA, HBW, 1);
  // bi-GRU layer 1: fwd/bwd -> hB halves
  gru_dir(hA, c_bwih + 2L * 384 * 256, bi_bih + 2 * 384, c_bwhh + 2L * 384 * 128,
          bi_bhh + 2 * 384, HBW, hB, 0, 0);
  gru_dir(hA, c_bwih + 3L * 384 * 256, bi_bih + 3 * 384, c_bwhh + 3L * 384 * 128,
          bi_bhh + 3 * 384, HBW, hB, HBW, 1);

  // attention: tmp = tanh(h @ w1^T + b1) [BT,128] f32 (reuse XG), scores, pool+BN
  float* TMP = XG;
  gemm(hB, HHW, c_aw1, HHW, att_b1, TMP, 128, BTN, 128, HHW, 1, 0);
  score_kernel<<<dim3(BTN / 256), dim3(256), 0, stream>>>(TMP, att_w2, att_b2, scores, BTN);
  attn_pool<<<dim3(BB), dim3(256), 0, stream>>>(hB, scores, bn_gamma, bn_beta, bn_mean,
                                                bn_var, ctx);
  // FC head (o1 kept bf16: it feeds the next WMMA GEMM)
  gemm(ctx, HHW, c_f1, HHW, fc_b1, o1, 128, BB, 128, HHW, 2, 1);
  gemm(o1, 128, c_f2, 128, fc_b2, o2, 64, BB, 64, 128, 2, 0);
  fc3_kernel<<<dim3(1), dim3(256), 0, stream>>>(o2, fc_w3, fc_b3, (float*)d_out);
}
